// CrossBlock_463856468491
// MI455X (gfx1250) — matrix operand order repaired and checked by reading, not compiled, not run on hardware
//
#include <hip/hip_runtime.h>
#include <math.h>

// ---------------------------------------------------------------------------
// CDNA5 (gfx1250) wave32 WMMA types / helpers
// ---------------------------------------------------------------------------
typedef __attribute__((ext_vector_type(16))) _Float16 v16h;
typedef __attribute__((ext_vector_type(8)))  float    v8f;

__device__ __forceinline__ v8f wmma_f32_16x16x32(v16h a, v16h b, v8f c) {
  // (neg_a, A, neg_b, B, c_mod, C, reuse_a, reuse_b)
  return __builtin_amdgcn_wmma_f32_16x16x32_f16(false, a, false, b, (short)0, c,
                                                false, false);
}

// Gather a 16x32 f16 A-fragment from a row-major LDS tile (ld = row stride).
// A layout (wave32): lane L -> row m = L%16, half = L/16;
// VGPR i holds K pair k = ((i>>2)<<4) | (half<<3) | ((i&3)<<1) | p  (p = 0/1).
__device__ __forceinline__ v16h ldsA_frag(const _Float16* base, int ld) {
  const int lane = threadIdx.x & 31;
  const int m = lane & 15, half = lane >> 4;
  union { v16h v; uint32_t u[8]; } r;
#pragma unroll
  for (int i = 0; i < 8; ++i) {
    const int k = ((i >> 2) << 4) | (half << 3) | ((i & 3) << 1);
    r.u[i] = *(const uint32_t*)(base + m * ld + k);
  }
  return r.v;
}

// ---------------------------------------------------------------------------
// Problem constants
// ---------------------------------------------------------------------------
#define BB 4
#define NN 2048
#define DD 256
#define HH 4
#define DH 64
#define RR (BB * NN)  // 8192 rows total
#define SSCALE 0.35355339059327373f  // DH^-0.25

// ---------------------------------------------------------------------------
// Kernel 1: convert fp32 weight [K, Nc] to f16 in WMMA B-fragment tile order.
// B layout (32x16 tile): lane -> col n = lane%16, half = lane/16;
// element e (v=e>>1, p=e&1) -> row k = ((v>>2)<<4) | (half<<3) | ((v&3)<<1) | p.
// ---------------------------------------------------------------------------
__global__ __launch_bounds__(256) void prep_weight(const float* __restrict__ src,
                                                   _Float16* __restrict__ dst,
                                                   int K, int Nc) {
  const int idx = blockIdx.x * 256 + threadIdx.x;  // exact: K*Nc threads
  const int tile = idx >> 9;
  const int within = idx & 511;
  const int lane = within >> 4, e = within & 15;
  const int ntiles = Nc >> 4;
  const int kt = tile / ntiles, nt = tile % ntiles;
  const int n = lane & 15, half = lane >> 4;
  const int v = e >> 1, p = e & 1;
  const int k = ((v >> 2) << 4) | (half << 3) | ((v & 3) << 1) | p;
  dst[idx] = (_Float16)src[(size_t)(kt * 32 + k) * Nc + (nt * 16 + n)];
}

// ---------------------------------------------------------------------------
// Kernel 2: projections for one stream. 32 rows per block, 8 waves.
// Waves 0-3: qk head h=w (scaled by SS); waves 4-7: v head h=w-4.
// Outputs (f16, pre-swizzled tiles per (b,h)):
//   qkA : [N,DH] A-tiles    (128 x 2 tiles)   -> flash Q operand
//   qkBt: [DH,N] B-tiles    (2 x 128 tiles)   -> flash K^T operand
//   vB  : [N,DH] B-tiles    (64 x 4 tiles)    -> flash V operand
// ---------------------------------------------------------------------------
__global__ __launch_bounds__(256) void proj_kernel(
    const float* __restrict__ x, const _Float16* __restrict__ Wqk,
    const _Float16* __restrict__ Wv, const float* __restrict__ bqk,
    const float* __restrict__ bv, _Float16* __restrict__ qkA,
    _Float16* __restrict__ qkBt, _Float16* __restrict__ vB) {
  __shared__ _Float16 Xs[32][256];     // 16 KB
  __shared__ _Float16 Ys[8][32][64];   // 32 KB (per-wave staging)
  const int tid = threadIdx.x;
  const int lane = tid & 31, w = tid >> 5;
  const int row0 = blockIdx.x * 32;

  for (int i = tid; i < 32 * 256; i += 256) {
    const int r = i >> 8, c = i & 255;
    Xs[r][c] = (_Float16)x[(size_t)(row0 + r) * 256 + c];
  }
  __syncthreads();

  const bool isV = (w >= 4);
  const int h = w & 3;
  const _Float16* Wt = isV ? Wv : Wqk;
  const float* bias = isV ? bv : bqk;
  const float scl = isV ? 1.0f : SSCALE;

  v8f acc[2][4] = {};
  for (int kt = 0; kt < 8; ++kt) {
    const v16h a0 = ldsA_frag(&Xs[0][kt * 32], 256);
    const v16h a1 = ldsA_frag(&Xs[16][kt * 32], 256);
#pragma unroll
    for (int nt = 0; nt < 4; ++nt) {
      const v16h bf =
          *(const v16h*)(Wt + (size_t)(kt * 16 + h * 4 + nt) * 512 + lane * 16);
      acc[0][nt] = wmma_f32_16x16x32(a0, bf, acc[0][nt]);
      acc[1][nt] = wmma_f32_16x16x32(a1, bf, acc[1][nt]);
    }
  }

  const int n = lane & 15, half = lane >> 4;
#pragma unroll
  for (int rt = 0; rt < 2; ++rt)
#pragma unroll
    for (int nt = 0; nt < 4; ++nt) {
      const float bval = bias[h * 64 + nt * 16 + n];
#pragma unroll
      for (int v = 0; v < 8; ++v)
        Ys[w][rt * 16 + v + 8 * half][nt * 16 + n] =
            (_Float16)((acc[rt][nt][v] + bval) * scl);
    }
  // Ys is per-wave: intra-wave DS ops are in order, no barrier needed.

  const int b = row0 >> 11;      // row0 / 2048
  const int n0 = row0 & 2047;    // seq offset within batch
  const size_t bh = (size_t)b * 4 + h;
  if (!isV) {
#pragma unroll
    for (int rt = 0; rt < 2; ++rt)
#pragma unroll
      for (int ct = 0; ct < 2; ++ct) {
        const v16h f = ldsA_frag(&Ys[w][rt * 16][ct * 32], 64);
        const size_t t = bh * 256 + (size_t)((n0 >> 4) + rt) * 2 + ct;
        *(v16h*)(qkA + t * 512 + lane * 16) = f;
      }
#pragma unroll
    for (int kt2 = 0; kt2 < 2; ++kt2)
#pragma unroll
      for (int ntt = 0; ntt < 2; ++ntt) {
        union { v16h v; uint32_t u[8]; } f;
#pragma unroll
        for (int i = 0; i < 8; ++i) {
          const int k = ((i >> 2) << 4) | (half << 3) | ((i & 3) << 1);  // dh index, contiguous pair
          f.u[i] = *(const uint32_t*)&Ys[w][ntt * 16 + n][kt2 * 32 + k];
        }
        const size_t t = bh * 256 + (size_t)kt2 * 128 + ((n0 >> 4) + ntt);
        *(v16h*)(qkBt + t * 512 + lane * 16) = f.v;
      }
  } else {
#pragma unroll
    for (int nt = 0; nt < 4; ++nt) {
      union { v16h v; _Float16 e[16]; } f;
#pragma unroll
      for (int i = 0; i < 8; ++i) {
        const int k = ((i >> 2) << 4) | (half << 3) | ((i & 3) << 1);  // seq index within 32
        f.e[2 * i]     = Ys[w][k][nt * 16 + n];
        f.e[2 * i + 1] = Ys[w][k + 1][nt * 16 + n];
      }
      const size_t t = bh * 256 + (size_t)(n0 >> 5) * 4 + nt;
      *(v16h*)(vB + t * 512 + lane * 16) = f.v;
    }
  }
}

// ---------------------------------------------------------------------------
// Kernel 3: flash attention, one wave per 16 query rows.
//   out[i, :] = softmax_j(Q[i]·K[j]) @ V      (per b,h)
// Launch twice: (qk0A, qk1Bt, v1B, m0A) and (qk1A, qk0Bt, v0B, m1A).
// Output: A-tiles of the merged-head [RR, DD] matrix (col = h*64 + dh).
// ---------------------------------------------------------------------------
__global__ __launch_bounds__(32) void flash_kernel(
    const _Float16* __restrict__ qA, const _Float16* __restrict__ kBt,
    const _Float16* __restrict__ vB, _Float16* __restrict__ mA) {
  __shared__ _Float16 Ps[16][32];
  __shared__ _Float16 Os[16][64];
  const int lane = threadIdx.x & 31;
  const int it = blockIdx.x;   // query row-tile 0..127
  const int bh = blockIdx.y;   // 0..15
  const size_t base = (size_t)bh * 256 * 512;
  const int n = lane & 15, half = lane >> 4;

  const v16h aq0 = *(const v16h*)(qA + base + ((size_t)it * 2 + 0) * 512 + lane * 16);
  const v16h aq1 = *(const v16h*)(qA + base + ((size_t)it * 2 + 1) * 512 + lane * 16);

  v8f o[4] = {};
  float rm[8], rl[8];
#pragma unroll
  for (int v = 0; v < 8; ++v) { rm[v] = -3.0e38f; rl[v] = 0.0f; }

  for (int jb = 0; jb < 64; ++jb) {
    v8f s0 = {}, s1 = {};
    {
      const v16h b00 = *(const v16h*)(kBt + base + ((size_t)jb * 2 + 0) * 512 + lane * 16);
      const v16h b01 = *(const v16h*)(kBt + base + ((size_t)jb * 2 + 1) * 512 + lane * 16);
      const v16h b10 = *(const v16h*)(kBt + base + ((size_t)128 + jb * 2 + 0) * 512 + lane * 16);
      const v16h b11 = *(const v16h*)(kBt + base + ((size_t)128 + jb * 2 + 1) * 512 + lane * 16);
      s0 = wmma_f32_16x16x32(aq0, b00, s0);
      s0 = wmma_f32_16x16x32(aq1, b10, s0);
      s1 = wmma_f32_16x16x32(aq0, b01, s1);
      s1 = wmma_f32_16x16x32(aq1, b11, s1);
    }
    float p0[8], p1[8];
#pragma unroll
    for (int v = 0; v < 8; ++v) {
      float bm = fmaxf(s0[v], s1[v]);
      bm = fmaxf(bm, __shfl_xor(bm, 1, 32));
      bm = fmaxf(bm, __shfl_xor(bm, 2, 32));
      bm = fmaxf(bm, __shfl_xor(bm, 4, 32));
      bm = fmaxf(bm, __shfl_xor(bm, 8, 32));
      const float nm = fmaxf(rm[v], bm);
      const float sc = __expf(rm[v] - nm);
      rm[v] = nm;
      p0[v] = __expf(s0[v] - nm);
      p1[v] = __expf(s1[v] - nm);
      float rs = p0[v] + p1[v];
      rs += __shfl_xor(rs, 1, 32);
      rs += __shfl_xor(rs, 2, 32);
      rs += __shfl_xor(rs, 4, 32);
      rs += __shfl_xor(rs, 8, 32);
      rl[v] = rl[v] * sc + rs;
#pragma unroll
      for (int d = 0; d < 4; ++d) o[d][v] *= sc;
    }
#pragma unroll
    for (int v = 0; v < 8; ++v) {
      Ps[v + 8 * half][n]      = (_Float16)p0[v];
      Ps[v + 8 * half][16 + n] = (_Float16)p1[v];
    }
    const v16h pa = ldsA_frag(&Ps[0][0], 32);  // intra-wave DS RAW: in order
#pragma unroll
    for (int d = 0; d < 4; ++d) {
      const v16h bv =
          *(const v16h*)(vB + base + ((size_t)jb * 4 + d) * 512 + lane * 16);
      o[d] = wmma_f32_16x16x32(pa, bv, o[d]);
    }
  }

#pragma unroll
  for (int d = 0; d < 4; ++d)
#pragma unroll
    for (int v = 0; v < 8; ++v)
      Os[v + 8 * half][d * 16 + n] = (_Float16)(o[d][v] / rl[v]);

  const int b = bh >> 2, h = bh & 3;
  const size_t rowtile = (size_t)b * 128 + it;
#pragma unroll
  for (int ct = 0; ct < 2; ++ct) {
    const v16h f = ldsA_frag(&Os[0][ct * 32], 64);
    *(v16h*)(mA + (rowtile * 8 + h * 2 + ct) * 512 + lane * 16) = f;
  }
}

// ---------------------------------------------------------------------------
// Kernel 4: epilogue for one stream. 16 rows per block, 8 waves.
//   mo = m@Wo + bo; cat = [x, mo]; h = cat@W1 + b1;
//   g = gelu(layernorm(h)); out = x + g@W2 + b2
// ---------------------------------------------------------------------------
__global__ __launch_bounds__(256) void epi_kernel(
    const float* __restrict__ x, const _Float16* __restrict__ mA,
    const _Float16* __restrict__ WoB, const float* __restrict__ bo,
    const _Float16* __restrict__ W1B, const float* __restrict__ b1,
    const float* __restrict__ ln_g, const float* __restrict__ ln_b,
    const _Float16* __restrict__ W2B, const float* __restrict__ b2,
    float* __restrict__ out) {
  __shared__ _Float16 cat[16][512];  // 16 KB
  __shared__ float hb[16][512];      // 32 KB
  __shared__ float mus[16], rvs[16];
  const int tid = threadIdx.x, lane = tid & 31, w = tid >> 5;
  const int rt = blockIdx.x;  // row tile 0..511
  const int row0 = rt * 16;
  const int n = lane & 15, half = lane >> 4;

  for (int i = tid; i < 16 * 256; i += 256) {
    const int r = i >> 8, c = i & 255;
    cat[r][c] = (_Float16)x[(size_t)(row0 + r) * 256 + c];
  }

  // mo = m @ Wo + bo  (wave w -> output cols [32w, 32w+32))
  v8f mo[2] = {};
  for (int kt = 0; kt < 8; ++kt) {
    const v16h a = *(const v16h*)(mA + ((size_t)rt * 8 + kt) * 512 + lane * 16);
#pragma unroll
    for (int q = 0; q < 2; ++q) {
      const int nt = 2 * w + q;
      const v16h bf = *(const v16h*)(WoB + (size_t)(kt * 16 + nt) * 512 + lane * 16);
      mo[q] = wmma_f32_16x16x32(a, bf, mo[q]);
    }
  }
#pragma unroll
  for (int q = 0; q < 2; ++q) {
    const int nt = 2 * w + q;
    const float bb = bo[nt * 16 + n];
#pragma unroll
    for (int v = 0; v < 8; ++v)
      cat[v + 8 * half][256 + nt * 16 + n] = (_Float16)(mo[q][v] + bb);
  }
  __syncthreads();

  // h = cat @ W1 + b1  (wave w -> cols [64w, 64w+64))
  v8f hh[4] = {};
  for (int kt = 0; kt < 16; ++kt) {
    const v16h a = ldsA_frag(&cat[0][kt * 32], 512);
#pragma unroll
    for (int q = 0; q < 4; ++q) {
      const int nt = 4 * w + q;
      const v16h bf = *(const v16h*)(W1B + (size_t)(kt * 32 + nt) * 512 + lane * 16);
      hh[q] = wmma_f32_16x16x32(a, bf, hh[q]);
    }
  }
#pragma unroll
  for (int q = 0; q < 4; ++q) {
    const int nt = 4 * w + q;
    const float bb = b1[nt * 16 + n];
#pragma unroll
    for (int v = 0; v < 8; ++v) hb[v + 8 * half][nt * 16 + n] = hh[q][v] + bb;
  }
  __syncthreads();

  if (tid < 16) {
    float s = 0.f;
    for (int c = 0; c < 512; ++c) s += hb[tid][c];
    const float mu = s * (1.0f / 512.0f);
    float vv = 0.f;
    for (int c = 0; c < 512; ++c) { const float d = hb[tid][c] - mu; vv += d * d; }
    mus[tid] = mu;
    rvs[tid] = rsqrtf(vv * (1.0f / 512.0f) + 1e-5f);
  }
  __syncthreads();

  for (int i = tid; i < 16 * 512; i += 256) {
    const int r = i >> 9, c = i & 511;
    const float vln = (hb[r][c] - mus[r]) * rvs[r] * ln_g[c] + ln_b[c];
    const float g = 0.5f * vln * (1.0f + erff(vln * 0.70710678118654752f));
    cat[r][c] = (_Float16)g;
  }
  __syncthreads();

  // out = x + g @ W2 + b2  (wave w -> cols [32w, 32w+32))
  v8f yy[2] = {};
  for (int kt = 0; kt < 16; ++kt) {
    const v16h a = ldsA_frag(&cat[0][kt * 32], 512);
#pragma unroll
    for (int q = 0; q < 2; ++q) {
      const int nt = 2 * w + q;
      const v16h bf = *(const v16h*)(W2B + (size_t)(kt * 16 + nt) * 512 + lane * 16);
      yy[q] = wmma_f32_16x16x32(a, bf, yy[q]);
    }
  }
#pragma unroll
  for (int q = 0; q < 2; ++q) {
    const int nt = 2 * w + q;
    const int col = nt * 16 + n;
    const float bb = b2[col];
#pragma unroll
    for (int v = 0; v < 8; ++v) {
      const int row = row0 + v + 8 * half;
      out[(size_t)row * 256 + col] =
          yy[q][v] + bb + x[(size_t)row * 256 + col];
    }
  }
}

// ---------------------------------------------------------------------------
// Host launcher
// ---------------------------------------------------------------------------
extern "C" void kernel_launch(void* const* d_in, const int* in_sizes, int n_in,
                              void* d_out, int out_size, void* d_ws, size_t ws_size,
                              hipStream_t stream) {
  (void)in_sizes; (void)n_in; (void)out_size; (void)ws_size;
  const float* x0  = (const float*)d_in[0];
  const float* x1  = (const float*)d_in[1];
  const float* Wqk = (const float*)d_in[2];
  const float* bqk = (const float*)d_in[3];
  const float* Wv  = (const float*)d_in[4];
  const float* bv  = (const float*)d_in[5];
  const float* Wo  = (const float*)d_in[6];
  const float* bo  = (const float*)d_in[7];
  const float* W1  = (const float*)d_in[8];
  const float* b1  = (const float*)d_in[9];
  const float* lg  = (const float*)d_in[10];
  const float* lb  = (const float*)d_in[11];
  const float* W2  = (const float*)d_in[12];
  const float* b2  = (const float*)d_in[13];
  float* out = (float*)d_out;

  char* ws = (char*)d_ws;
  size_t off = 0;
  auto take = [&](size_t bytes) { char* p = ws + off; off += bytes; return p; };
  _Float16* WqkB  = (_Float16*)take(131072);    // 256x256 f16 B-tiles
  _Float16* WvB   = (_Float16*)take(131072);
  _Float16* WoB   = (_Float16*)take(131072);
  _Float16* W1B   = (_Float16*)take(524288);    // 512x512
  _Float16* W2B   = (_Float16*)take(262144);    // 512x256
  _Float16* qk0A  = (_Float16*)take(4194304);
  _Float16* qk1A  = (_Float16*)take(4194304);
  _Float16* qk0Bt = (_Float16*)take(4194304);
  _Float16* qk1Bt = (_Float16*)take(4194304);
  _Float16* v0B   = (_Float16*)take(4194304);
  _Float16* v1B   = (_Float16*)take(4194304);
  _Float16* m0A   = (_Float16*)take(4194304);
  _Float16* m1A   = (_Float16*)take(4194304);

  prep_weight<<<256, 256, 0, stream>>>(Wqk, WqkB, 256, 256);
  prep_weight<<<256, 256, 0, stream>>>(Wv,  WvB,  256, 256);
  prep_weight<<<256, 256, 0, stream>>>(Wo,  WoB,  256, 256);
  prep_weight<<<1024, 256, 0, stream>>>(W1, W1B,  512, 512);
  prep_weight<<<512, 256, 0, stream>>>(W2,  W2B,  512, 256);

  proj_kernel<<<256, 256, 0, stream>>>(x0, WqkB, WvB, bqk, bv, qk0A, qk0Bt, v0B);
  proj_kernel<<<256, 256, 0, stream>>>(x1, WqkB, WvB, bqk, bv, qk1A, qk1Bt, v1B);

  flash_kernel<<<dim3(128, 16), 32, 0, stream>>>(qk0A, qk1Bt, v1B, m0A);
  flash_kernel<<<dim3(128, 16), 32, 0, stream>>>(qk1A, qk0Bt, v0B, m1A);

  epi_kernel<<<512, 256, 0, stream>>>(x0, m0A, WoB, bo, W1B, b1, lg, lb, W2B, b2,
                                      out);
  epi_kernel<<<512, 256, 0, stream>>>(x1, m1A, WoB, bo, W1B, b1, lg, lb, W2B, b2,
                                      out + (size_t)RR * DD);
}
